// MAB_83494164234881
// MI455X (gfx1250) — matrix operand order repaired and checked by reading, not compiled, not run on hardware
//
#include <hip/hip_runtime.h>
#include <hip/hip_bf16.h>
#include <stdint.h>

typedef _Float16 f16;
typedef _Float16 v16h __attribute__((ext_vector_type(16)));
typedef float    v8f  __attribute__((ext_vector_type(8)));

union Frag16 { v16h h; uint4 u[2]; };
union Half8  { uint4 u; f16 h[8]; };

__device__ __forceinline__ v8f zero8() {
  v8f z;
#pragma unroll
  for (int i = 0; i < 8; ++i) z[i] = 0.0f;
  return z;
}

// xor-lane shuffle via DS_SWIZZLE with immediate pattern (group-of-32 mode:
// new_lane = (lane & 0x1f) ^ XOR). No index VGPR, no bpermute setup.
template <int XOR>
__device__ __forceinline__ float swz_xor(float x) {
  return __int_as_float(
      __builtin_amdgcn_ds_swizzle(__float_as_int(x), (XOR << 10) | 0x1f));
}

// batched 8-row max reduction: issue all 8 swizzles, then combine, so the
// in-order DS returns pipeline instead of serializing on s_wait_dscnt 0.
template <int XOR>
__device__ __forceinline__ void redmax8(float* m) {
  float t[8];
#pragma unroll
  for (int v = 0; v < 8; ++v) t[v] = swz_xor<XOR>(m[v]);
#pragma unroll
  for (int v = 0; v < 8; ++v) m[v] = fmaxf(m[v], t[v]);
}

#define FEAT 1024   // DQ = DK = DV
#define NTOK 1024   // NQ = NK
#define NBAT 4
#define NHEAD 16
#define HDIM 64

// ---------------------------------------------------------------------------
// Kernel: 32x32 tiled transpose + f32 -> f16 convert (weights -> BT layout)
// ---------------------------------------------------------------------------
__global__ __launch_bounds__(256) void k_transpose_h(const float* __restrict__ w,
                                                     f16* __restrict__ wT) {
  __shared__ float tile[32][33];
  const int bx = blockIdx.x * 32, by = blockIdx.y * 32;
  const int tx = threadIdx.x, ty = threadIdx.y;  // 32 x 8
#pragma unroll
  for (int i = 0; i < 32; i += 8)
    tile[ty + i][tx] = w[(size_t)(by + ty + i) * FEAT + bx + tx];
  __syncthreads();
#pragma unroll
  for (int i = 0; i < 32; i += 8)
    wT[(size_t)(bx + ty + i) * FEAT + by + tx] = (f16)tile[tx][ty + i];
}

// ---------------------------------------------------------------------------
// Kernel: row LayerNorm (1024 features) -> f16 output. One block per row.
// ---------------------------------------------------------------------------
__global__ __launch_bounds__(256) void k_ln_f16(const float* __restrict__ x,
                                                const float* __restrict__ g,
                                                const float* __restrict__ be,
                                                f16* __restrict__ out) {
  const int row = blockIdx.x;
  const float* xr = x + (size_t)row * FEAT;
  float v0[4];
  float s = 0.0f, s2 = 0.0f;
#pragma unroll
  for (int i = 0; i < 4; ++i) {
    float v = xr[threadIdx.x + i * 256];
    v0[i] = v; s += v; s2 += v * v;
  }
  s += swz_xor<1>(s);   s2 += swz_xor<1>(s2);
  s += swz_xor<2>(s);   s2 += swz_xor<2>(s2);
  s += swz_xor<4>(s);   s2 += swz_xor<4>(s2);
  s += swz_xor<8>(s);   s2 += swz_xor<8>(s2);
  s += swz_xor<16>(s);  s2 += swz_xor<16>(s2);
  __shared__ float ps[8], ps2[8];
  const int wid = threadIdx.x >> 5, lane = threadIdx.x & 31;
  if (lane == 0) { ps[wid] = s; ps2[wid] = s2; }
  __syncthreads();
  if (wid == 0) {
    float a = (lane < 8) ? ps[lane]  : 0.0f;
    float b = (lane < 8) ? ps2[lane] : 0.0f;
    a += swz_xor<1>(a);  b += swz_xor<1>(b);
    a += swz_xor<2>(a);  b += swz_xor<2>(b);
    a += swz_xor<4>(a);  b += swz_xor<4>(b);
    if (lane == 0) { ps[0] = a; ps2[0] = b; }
  }
  __syncthreads();
  const float mean = ps[0] * (1.0f / FEAT);
  const float var  = ps2[0] * (1.0f / FEAT) - mean * mean;
  const float rstd = rsqrtf(var + 1e-5f);
#pragma unroll
  for (int i = 0; i < 4; ++i) {
    const int c = threadIdx.x + i * 256;
    out[(size_t)row * FEAT + c] = (f16)((v0[i] - mean) * rstd * g[c] + be[c]);
  }
}

// ---------------------------------------------------------------------------
// Kernel: f16 GEMM  C[M,N] = A[M,K] * B[K,N] + bias, with B given transposed
// (BT[N,K] row-major).  128x128 block tile, 8 waves of 64x32, LDS stage of
// K=64 (2 wmma k-steps per barrier pair), v_wmma_f32_16x16x32_f16.
//   EPI == 0 : outH = (f16)(acc + bias)
//   EPI == 1 : outF = res + relu(acc + bias)      (fp32)
// ---------------------------------------------------------------------------
template <int EPI>
__global__ __launch_bounds__(256) void k_gemm_wmma(
    const f16* __restrict__ A, const f16* __restrict__ BT,
    const float* __restrict__ bias, const float* __restrict__ res,
    f16* __restrict__ outH, float* __restrict__ outF) {
  constexpr int KD = FEAT;        // K dim / leading dim
  constexpr int ND = FEAT;        // output leading dim
  constexpr int LDS_STRIDE = 72;  // 64 halves + 8 pad (bank spread)

  __shared__ f16 sA[128 * LDS_STRIDE];   // 18 KB
  __shared__ f16 sB[128 * LDS_STRIDE];   // 18 KB

  const int bm = blockIdx.y, bn = blockIdx.x;
  const int tid = threadIdx.x;
  const int wid = tid >> 5, lane = tid & 31;
  const int wm = wid & 1, wn = wid >> 1;          // 2 x 4 wave grid
  const int l16 = lane & 15, lh = lane >> 4;

  v8f acc[4][2];
#pragma unroll
  for (int i = 0; i < 4; ++i)
#pragma unroll
    for (int j = 0; j < 2; ++j) acc[i][j] = zero8();

  const int ldr = tid >> 1;             // 0..127 tile row for loads
  const int ldc = (tid & 1) * 32;       // half-offset within 64-wide k slice

  for (int kk = 0; kk < KD; kk += 64) {
    // --- stage A/B 128x64 tiles into LDS (4x uint4 per thread per matrix) ---
    const uint4* ga = (const uint4*)(A  + (size_t)(bm * 128 + ldr) * KD + kk + ldc);
    const uint4* gb = (const uint4*)(BT + (size_t)(bn * 128 + ldr) * KD + kk + ldc);
#pragma unroll
    for (int q = 0; q < 4; ++q) {
      *(uint4*)(sA + ldr * LDS_STRIDE + ldc + q * 8) = ga[q];
      *(uint4*)(sB + ldr * LDS_STRIDE + ldc + q * 8) = gb[q];
    }
    if (kk + 64 < KD) {   // hint next tiles toward the cache hierarchy
      __builtin_prefetch(A  + (size_t)(bm * 128 + ldr) * KD + kk + 64 + ldc, 0, 1);
      __builtin_prefetch(BT + (size_t)(bn * 128 + ldr) * KD + kk + 64 + ldc, 0, 1);
    }
    __syncthreads();

#pragma unroll
    for (int ks = 0; ks < 2; ++ks) {
      const int kc = ks * 32;
      // --- B fragments: lanes 0-15 hold K=0..15, lanes 16-31 hold K=16..31 ---
      v16h bfrag[2];
#pragma unroll
      for (int j = 0; j < 2; ++j) {
        const f16* p = sB + (wn * 32 + j * 16 + l16) * LDS_STRIDE + kc + lh * 16;
        Frag16 fb;
        fb.u[0] = *(const uint4*)p;
        fb.u[1] = *(const uint4*)(p + 8);
        bfrag[j] = fb.h;
      }
      // --- A fragments: lanes 0-15 hold K=0..15, lanes 16-31 hold K=16..31 ---
#pragma unroll
      for (int i = 0; i < 4; ++i) {
        const f16* p = sA + (wm * 64 + i * 16 + l16) * LDS_STRIDE + kc;
        Frag16 fa;
        fa.u[0] = *(const uint4*)(p + lh * 16);
        fa.u[1] = *(const uint4*)(p + 8 + lh * 16);
        const v16h af = fa.h;
#pragma unroll
        for (int j = 0; j < 2; ++j)
          acc[i][j] = __builtin_amdgcn_wmma_f32_16x16x32_f16(
              false, af, false, bfrag[j], (short)0, acc[i][j], false, false);
      }
    }
    __syncthreads();
  }

  // --- epilogue; C layout: lane(v): M = v + 8*(lane>=16), N = lane&15 ---
  const int row0 = bm * 128 + wm * 64;
  const int col0 = bn * 128 + wn * 32 + l16;
#pragma unroll
  for (int j = 0; j < 2; ++j) {
    const int c = col0 + j * 16;
    const float bv = bias[c];
#pragma unroll
    for (int i = 0; i < 4; ++i) {
#pragma unroll
      for (int v = 0; v < 8; ++v) {
        const int r = row0 + i * 16 + v + lh * 8;
        const float val = acc[i][j][v] + bv;
        if (EPI == 0) {
          outH[(size_t)r * ND + c] = (f16)val;
        } else {
          const float rv = res[(size_t)r * ND + c];
          outF[(size_t)r * ND + c] = rv + (val > 0.0f ? val : 0.0f);
        }
      }
    }
  }
}

// ---------------------------------------------------------------------------
// Kernel: flash attention per (128 q-rows, head, batch).  256 threads = 8 waves,
// each wave owns 16 q-rows; K/V tile staging amortized over all 8 waves.
// Online softmax: row max via batched ds_swizzle; row sum via a WMMA against an
// all-ones B fragment (l += P @ 1), so no cross-lane sum traffic at all.
// Writes O = Q_residual + ctx (fp32).
// ---------------------------------------------------------------------------
__global__ __launch_bounds__(256) void k_attn(const f16* __restrict__ Qp,
                                              const f16* __restrict__ Kp,
                                              const f16* __restrict__ Vp,
                                              const float* __restrict__ Qin,
                                              float* __restrict__ O) {
  constexpr int LS = 72;  // padded LDS row stride (halves)
  __shared__ f16 sK[64 * LS];      // K tile, rows = kv token, cols = d (BT for Q*K^T)
  __shared__ f16 sVT[64 * LS];     // V tile transposed (BT for P*V)
  __shared__ f16 sP[8 * 16 * LS];  // per-wave P staging (16 rows x 64)

  const int qb = blockIdx.x * 128, h = blockIdx.y, b = blockIdx.z;
  const int tid = threadIdx.x;
  const int wid = tid >> 5, lane = tid & 31;
  const int l16 = lane & 15, lh = lane >> 4;

  // preload this wave's Q fragments, pre-scaled by 1/sqrt(DV) = 1/32 (exact in f16)
  const size_t qrow0 = (size_t)b * NTOK + qb + wid * 16;
  v16h qfrag[2];
#pragma unroll
  for (int s = 0; s < 2; ++s) {
    const f16* p = Qp + (qrow0 + l16) * FEAT + h * HDIM + s * 32;
    Frag16 f;
    f.u[0] = *(const uint4*)(p + lh * 16);
    f.u[1] = *(const uint4*)(p + 8 + lh * 16);
    v16h q = f.h;
#pragma unroll
    for (int e = 0; e < 16; ++e) q[e] = q[e] * (f16)0.03125f;
    qfrag[s] = q;
  }

  // all-ones B fragment: P @ ones gives per-row sums in accumulator layout
  v16h bones;
#pragma unroll
  for (int e = 0; e < 16; ++e) bones[e] = (f16)1.0f;

  float mrow[8], lrow[8];
#pragma unroll
  for (int v = 0; v < 8; ++v) { mrow[v] = -3.0e38f; lrow[v] = 0.0f; }
  v8f oacc[4];
#pragma unroll
  for (int n = 0; n < 4; ++n) oacc[n] = zero8();

  for (int kt = 0; kt < NTOK / 64; ++kt) {
    const int kr0 = kt * 64;
    // --- stage K tile row-major (already BT layout for scores) ---
    for (int i = tid; i < 64 * 8; i += 256) {
      const int r = i >> 3, c8 = (i & 7) * 8;
      *(uint4*)(sK + r * LS + c8) =
          *(const uint4*)(Kp + ((size_t)b * NTOK + kr0 + r) * FEAT + h * HDIM + c8);
    }
    // --- stage V tile transposed: coalesced uint4 row loads, b16 scatter ---
    for (int i = tid; i < 64 * 8; i += 256) {
      const int r = i >> 3, c8 = (i & 7) * 8;
      Half8 hv;
      hv.u = *(const uint4*)(Vp + ((size_t)b * NTOK + kr0 + r) * FEAT + h * HDIM + c8);
#pragma unroll
      for (int e = 0; e < 8; ++e) sVT[(c8 + e) * LS + r] = hv.h[e];
    }
    __syncthreads();

    // --- scores S[16 x 64] = (Q/32) * K^T : 4 n-frags x 2 k-steps ---
    v8f sacc[4];
#pragma unroll
    for (int nf = 0; nf < 4; ++nf) {
      sacc[nf] = zero8();
#pragma unroll
      for (int ks = 0; ks < 2; ++ks) {
        const f16* p = sK + (nf * 16 + l16) * LS + ks * 32 + lh * 16;
        Frag16 fb;
        fb.u[0] = *(const uint4*)p;
        fb.u[1] = *(const uint4*)(p + 8);
        sacc[nf] = __builtin_amdgcn_wmma_f32_16x16x32_f16(
            false, qfrag[ks], false, fb.h, (short)0, sacc[nf], false, false);
      }
    }

    // --- online softmax (rows = v + 8*lh, each row spread over 16 lanes) ---
    float mx[8];
#pragma unroll
    for (int v = 0; v < 8; ++v)
      mx[v] = fmaxf(fmaxf(sacc[0][v], sacc[1][v]), fmaxf(sacc[2][v], sacc[3][v]));
    redmax8<1>(mx); redmax8<2>(mx); redmax8<4>(mx); redmax8<8>(mx);

    float alpha[8];
#pragma unroll
    for (int v = 0; v < 8; ++v) {
      const float mnew = fmaxf(mrow[v], mx[v]);
      alpha[v] = __expf(mrow[v] - mnew);
      mrow[v] = mnew;
#pragma unroll
      for (int nf = 0; nf < 4; ++nf) sacc[nf][v] = __expf(sacc[nf][v] - mnew);
      lrow[v] *= alpha[v];
#pragma unroll
      for (int nf = 0; nf < 4; ++nf) oacc[nf][v] *= alpha[v];
      // restage P as f16 in row-major per-wave LDS (accumulator -> A layout)
      f16* pp = sP + wid * 16 * LS + (v + lh * 8) * LS;
#pragma unroll
      for (int nf = 0; nf < 4; ++nf) pp[nf * 16 + l16] = (f16)sacc[nf][v];
    }

    // --- ctx += P * V, and lrow += P @ ones (row-sum via WMMA) ---
    v16h pfrag[2];
#pragma unroll
    for (int ks = 0; ks < 2; ++ks) {
      const f16* p = sP + wid * 16 * LS + l16 * LS + ks * 32;
      Frag16 f;
      f.u[0] = *(const uint4*)(p + lh * 16);
      f.u[1] = *(const uint4*)(p + 8 + lh * 16);
      pfrag[ks] = f.h;
    }
    v8f racc = zero8();
#pragma unroll
    for (int ks = 0; ks < 2; ++ks)
      racc = __builtin_amdgcn_wmma_f32_16x16x32_f16(
          false, pfrag[ks], false, bones, (short)0, racc, false, false);
#pragma unroll
    for (int nf = 0; nf < 4; ++nf) {
#pragma unroll
      for (int ks = 0; ks < 2; ++ks) {
        const f16* p = sVT + (nf * 16 + l16) * LS + ks * 32 + lh * 16;
        Frag16 fb;
        fb.u[0] = *(const uint4*)p;
        fb.u[1] = *(const uint4*)(p + 8);
        oacc[nf] = __builtin_amdgcn_wmma_f32_16x16x32_f16(
            false, pfrag[ks], false, fb.h, (short)0, oacc[nf], false, false);
      }
    }
#pragma unroll
    for (int v = 0; v < 8; ++v) lrow[v] += racc[v];
    __syncthreads();
  }

  // --- finalize: O = Q + ctx / l ---
#pragma unroll
  for (int nf = 0; nf < 4; ++nf) {
    const int c = h * HDIM + nf * 16 + l16;
#pragma unroll
    for (int v = 0; v < 8; ++v) {
      const int r = qb + wid * 16 + v + lh * 8;
      const size_t idx = ((size_t)b * NTOK + r) * FEAT + c;
      O[idx] = Qin[idx] + oacc[nf][v] * (1.0f / lrow[v]);
    }
  }
}

// ---------------------------------------------------------------------------
// Host-side launch
// ---------------------------------------------------------------------------
extern "C" void kernel_launch(void* const* d_in, const int* in_sizes, int n_in,
                              void* d_out, int out_size, void* d_ws, size_t ws_size,
                              hipStream_t stream) {
  const float* Q  = (const float*)d_in[0];
  const float* K  = (const float*)d_in[1];
  const float* wq = (const float*)d_in[2];
  const float* bq = (const float*)d_in[3];
  const float* wk = (const float*)d_in[4];
  const float* bk = (const float*)d_in[5];
  const float* wv = (const float*)d_in[6];
  const float* bv = (const float*)d_in[7];
  const float* wo = (const float*)d_in[8];
  const float* bo = (const float*)d_in[9];
  const float* gq = (const float*)d_in[10];
  const float* betaq = (const float*)d_in[11];
  const float* gk = (const float*)d_in[12];
  const float* betak = (const float*)d_in[13];
  const float* g0 = (const float*)d_in[14];
  const float* beta0 = (const float*)d_in[15];
  float* out = (float*)d_out;

  const size_t MB = 1u << 20;
  // workspace layout (bytes)
  char* ws = (char*)d_ws;
  f16* QnH = (f16*)(ws + 0 * MB);    // 4096x1024 f16  (8 MB)
  f16* KnH = (f16*)(ws + 8 * MB);    // 8 MB
  f16* wqT = (f16*)(ws + 16 * MB);   // 2 MB
  f16* wkT = (f16*)(ws + 18 * MB);
  f16* wvT = (f16*)(ws + 20 * MB);
  f16* woT = (f16*)(ws + 22 * MB);
  f16* QpH = (f16*)(ws + 24 * MB);   // 8 MB
  f16* KpH = (f16*)(ws + 32 * MB);   // 8 MB
  f16* VpH = (f16*)(ws + 40 * MB);   // 8 MB
  float* Obuf = (float*)(ws + 48 * MB);  // 16 MB
  f16* OnH = (f16*)(ws + 64 * MB);   // 8 MB
  if (ws_size < 72 * MB) return;

  const int M = NBAT * NTOK;  // 4096 rows

  // 1) weight transpose + convert
  {
    dim3 g(FEAT / 32, FEAT / 32), blk(32, 8);
    k_transpose_h<<<g, blk, 0, stream>>>(wq, wqT);
    k_transpose_h<<<g, blk, 0, stream>>>(wk, wkT);
    k_transpose_h<<<g, blk, 0, stream>>>(wv, wvT);
    k_transpose_h<<<g, blk, 0, stream>>>(wo, woT);
  }
  // 2) pre-norms -> f16
  k_ln_f16<<<M, 256, 0, stream>>>(Q, gq, betaq, QnH);
  k_ln_f16<<<M, 256, 0, stream>>>(K, gk, betak, KnH);
  // 3) projections (f16 out)
  {
    dim3 g(FEAT / 128, M / 128), blk(256);
    k_gemm_wmma<0><<<g, blk, 0, stream>>>(QnH, wqT, bq, nullptr, QpH, nullptr);
    k_gemm_wmma<0><<<g, blk, 0, stream>>>(KnH, wkT, bk, nullptr, KpH, nullptr);
    k_gemm_wmma<0><<<g, blk, 0, stream>>>(KnH, wvT, bv, nullptr, VpH, nullptr);
  }
  // 4) attention + residual -> Obuf (fp32)
  {
    dim3 g(NTOK / 128, NHEAD, NBAT), blk(256);
    k_attn<<<g, blk, 0, stream>>>(QpH, KpH, VpH, Q, Obuf);
  }
  // 5) post-norm -> f16
  k_ln_f16<<<M, 256, 0, stream>>>(Obuf, g0, beta0, OnH);
  // 6) output GEMM with ReLU + residual -> d_out (fp32)
  {
    dim3 g(FEAT / 128, M / 128), blk(256);
    k_gemm_wmma<1><<<g, blk, 0, stream>>>(OnH, woT, bo, Obuf, nullptr, out);
  }
}
